// NeuralCDEEncoder_44805098832349
// MI455X (gfx1250) — compile-verified, matrix operand order checked by reading, not run on hardware
//
#include <hip/hip_runtime.h>

// Problem constants (match reference)
#define Bb 64
#define Ll 128
#define Dd 64
#define Hh 768
#define Oo 256
#define NSTEPS 20

typedef _Float16 v16h __attribute__((ext_vector_type(16)));
typedef _Float16 half8 __attribute__((ext_vector_type(8)));
typedef _Float16 half4 __attribute__((ext_vector_type(4)));
typedef float    v8f  __attribute__((ext_vector_type(8)));
typedef float    f8   __attribute__((ext_vector_type(8)));
typedef float    f4   __attribute__((ext_vector_type(4)));

// ---------------------------------------------------------------------------
// K1: spline coefficients. dX(b,d,t) = p[b,d] + q[b,d]*t^2  (t in [0,1) =>
// spline interval 0 only). Reverse Thomas sweep gives M1 (2nd deriv at knot 1)
// with O(1) state: M_i = g_i - h_i*M_{i-1}, M_0_interior = g_0 since M_{-1}=0.
// p,q stored transposed (D,B) so the step kernel can load 8 batches contiguously.
// ---------------------------------------------------------------------------
__global__ __launch_bounds__(256) void spline_kernel(
    const float* __restrict__ traj, float* __restrict__ pT, float* __restrict__ qT)
{
  int tid = blockIdx.x * blockDim.x + threadIdx.x;
  if (tid >= Bb * Dd) return;
  int b = tid / Dd, d = tid % Dd;
  const float* y = traj + (size_t)b * Ll * Dd + d;   // stride Dd over l
  float w2 = y[(Ll - 1) * Dd];
  float w1 = y[(Ll - 2) * Dd];
  float g = 0.f, h = 0.f;
  for (int i = Ll - 3; i >= 0; --i) {                // interior rows 125..0
    float w0 = y[i * Dd];
    float r  = 6.f * (w2 - 2.f * w1 + w0);
    float inv = 1.f / (4.f - h);
    g = (r - g) * inv;
    h = inv;
    w2 = w1; w1 = w0;
  }
  float M1 = g;                                      // w1=y0, w2=y1 now
  pT[d * Bb + b] = (w2 - w1) - M1 * (1.f / 6.f);     // b0 coefficient
  qT[d * Bb + b] = M1 * 0.5f;                        // 3*d0 coefficient
}

// ---------------------------------------------------------------------------
// fp32 -> f16 conversion (vectorized grid-stride)
// ---------------------------------------------------------------------------
__global__ __launch_bounds__(256) void cvt_kernel(
    const float* __restrict__ src, _Float16* __restrict__ dst, int n4)
{
  int i = blockIdx.x * blockDim.x + threadIdx.x;
  int stride = gridDim.x * blockDim.x;
  for (; i < n4; i += stride) {
    f4 v = ((const f4*)src)[i];
    half4 o;
    #pragma unroll
    for (int j = 0; j < 4; ++j) o[j] = (_Float16)v[j];
    ((half4*)dst)[i] = o;
  }
}

// ---------------------------------------------------------------------------
// K2: z0 = traj[:,0,:] @ W_z0^T + b_z0   (tiny: 64x64x768)
// ---------------------------------------------------------------------------
__global__ __launch_bounds__(256) void z0_kernel(
    const float* __restrict__ traj, const float* __restrict__ Wz0,
    const float* __restrict__ bz0,
    float* __restrict__ z32, _Float16* __restrict__ z16, float* __restrict__ zs0)
{
  int tid = blockIdx.x * blockDim.x + threadIdx.x;
  if (tid >= Bb * Hh) return;
  int b = tid / Hh, h = tid % Hh;
  const float* x = traj + (size_t)b * Ll * Dd;       // row l=0
  const float* w = Wz0 + (size_t)h * Dd;
  float s = bz0[h];
  #pragma unroll 8
  for (int d = 0; d < Dd; ++d) s += x[d] * w[d];
  z32[tid] = s;
  z16[tid] = (_Float16)s;
  zs0[tid] = s;
}

// ---------------------------------------------------------------------------
// K3: one Euler step.
//   G[b,n] = sum_j z[b,j] * W_lin[n,j]          (n = h*64+d), f16 WMMA, f32 acc
//   z_new[b,h] = z[b,h] + dt * sum_d (G[b,h*64+d] + b_lin[h*64+d]) * dX[b,d]
// One wave per (btile, h): 4 accumulators cover d=0..63, K-loop over 768.
// Wave mapping: btile = wid & 3, h = wid >> 2  =>  the 4 waves of one block
// share the SAME h (same W_lin rows), so B-fragment loads hit the WGP$ 3/4 of
// the time; W_lin f16 (75.5 MB) streams from L2 exactly once per step.
// A frag: lane l holds row (l&15); halfs e<8 -> K = kb+e, e>=8 -> K = kb+16+(e-8),
// kb = (l>=16)*8  => two contiguous b128 loads. B mirrors A with col = (l&15),
// and W_lin rows are contiguous over K, so the same two-b128 pattern applies.
// C frag: lane l holds col (l&15); VGPR v holds row m = v + 8*(l>=16).
// ---------------------------------------------------------------------------
__global__ __launch_bounds__(128) void step_kernel(
    const float* __restrict__ zin, const _Float16* __restrict__ zinH,
    const _Float16* __restrict__ Wh, const float* __restrict__ b_lin,
    const float* __restrict__ pT, const float* __restrict__ qT,
    float t, float dt,
    float* __restrict__ zout, _Float16* __restrict__ zoutH,
    float* __restrict__ zs_slot)
{
  const int lane  = threadIdx.x & 31;
  const int wid   = (blockIdx.x * blockDim.x + threadIdx.x) >> 5;   // 0..3071
  const int btile = wid & 3;                                        // 0..3
  const int h     = wid >> 2;                                       // 0..767
  const int half  = lane >> 4;
  const int l16   = lane & 15;
  const int kb    = half * 8;

  const int arow = btile * 16 + l16;                                // batch row
  const _Float16* Abase = zinH + (size_t)arow * Hh;

  v8f acc[4] = {};

  for (int k0 = 0; k0 < Hh; k0 += 32) {
    v16h a;
    {
      half8 lo = *(const half8*)(Abase + k0 + kb);
      half8 hi = *(const half8*)(Abase + k0 + kb + 16);
      #pragma unroll
      for (int i = 0; i < 8; ++i) { a[i] = lo[i]; a[8 + i] = hi[i]; }
    }
    #pragma unroll
    for (int w = 0; w < 4; ++w) {
      const int n = h * Dd + w * 16 + l16;                          // W_lin row
      const _Float16* Bbase = Wh + (size_t)n * Hh + k0 + kb;
      if (k0 + 32 < Hh) __builtin_prefetch((const void*)(Bbase + 32), 0, 1);
      half8 lo = *(const half8*)(Bbase);
      half8 hi = *(const half8*)(Bbase + 16);
      v16h bf;
      #pragma unroll
      for (int i = 0; i < 8; ++i) { bf[i] = lo[i]; bf[8 + i] = hi[i]; }
      acc[w] = __builtin_amdgcn_wmma_f32_16x16x32_f16(
          false, a, false, bf, (short)0, acc[w], false, false);
    }
  }

  // Weighted reduction over d: dX = p + q*t^2 (spline interval 0 only)
  const float t2 = t * t;
  const int b0 = btile * 16 + half * 8;     // first batch in this lane's m-range
  float tv[8];
  #pragma unroll
  for (int v = 0; v < 8; ++v) tv[v] = 0.f;
  #pragma unroll
  for (int w = 0; w < 4; ++w) {
    const int d = w * 16 + l16;
    const float bl = b_lin[h * Dd + d];
    const f8 pv = *(const f8*)(pT + d * Bb + b0);
    const f8 qv = *(const f8*)(qT + d * Bb + b0);
    #pragma unroll
    for (int v = 0; v < 8; ++v) {
      float dx = pv[v] + qv[v] * t2;
      tv[v] += (acc[w][v] + bl) * dx;
    }
  }
  // butterfly sum across the 16 lanes of each half-wave (all share same m-range)
  #pragma unroll
  for (int m = 1; m < 16; m <<= 1) {
    #pragma unroll
    for (int v = 0; v < 8; ++v) tv[v] += __shfl_xor(tv[v], m, 16);
  }

  if (l16 == 0) {
    #pragma unroll
    for (int v = 0; v < 8; ++v) {
      const int b   = b0 + v;
      const int idx = b * Hh + h;
      float zn = zin[idx] + dt * tv[v];
      zout[idx]    = zn;
      zoutH[idx]   = (_Float16)zn;
      zs_slot[idx] = zn;
    }
  }
}

// ---------------------------------------------------------------------------
// K4: project all grid states: y[k,b,o] = zs[k,b,:] @ W_out[o,:] + b_out[o]
// (21*64 = 1344 rows) x K=768 x N=256, one wave per 16x16 output tile.
// ---------------------------------------------------------------------------
__global__ __launch_bounds__(128) void outproj_kernel(
    const _Float16* __restrict__ Ah,    // (1344, 768)
    const _Float16* __restrict__ Wh,    // (256, 768)
    const float* __restrict__ b_out,
    float* __restrict__ y)              // (1344, 256)
{
  const int lane = threadIdx.x & 31;
  const int wid  = (blockIdx.x * blockDim.x + threadIdx.x) >> 5;    // 0..1343
  const int mt   = wid >> 4;                                        // 0..83
  const int nt   = wid & 15;                                        // 0..15
  const int half = lane >> 4;
  const int l16  = lane & 15;
  const int kb   = half * 8;
  const int arow = mt * 16 + l16;
  const int ncol = nt * 16 + l16;
  const _Float16* Abase = Ah + (size_t)arow * Hh;
  const _Float16* Bbase = Wh + (size_t)ncol * Hh;
  v8f acc = {};
  for (int k0 = 0; k0 < Hh; k0 += 32) {
    half8 alo = *(const half8*)(Abase + k0 + kb);
    half8 ahi = *(const half8*)(Abase + k0 + kb + 16);
    half8 blo = *(const half8*)(Bbase + k0 + kb);
    half8 bhi = *(const half8*)(Bbase + k0 + kb + 16);
    v16h a, bf;
    #pragma unroll
    for (int i = 0; i < 8; ++i) {
      a[i] = alo[i]; a[8 + i] = ahi[i];
      bf[i] = blo[i]; bf[8 + i] = bhi[i];
    }
    acc = __builtin_amdgcn_wmma_f32_16x16x32_f16(
        false, a, false, bf, (short)0, acc, false, false);
  }
  const float bo = b_out[ncol];
  const int m0 = mt * 16 + half * 8;
  #pragma unroll
  for (int v = 0; v < 8; ++v)
    y[(size_t)(m0 + v) * Oo + ncol] = acc[v] + bo;
}

// ---------------------------------------------------------------------------
// K5: interpolate grid outputs onto ts = linspace(0,1,L)
// ---------------------------------------------------------------------------
__global__ __launch_bounds__(256) void interp_kernel(
    const float* __restrict__ y, float* __restrict__ out)
{
  int tid = blockIdx.x * blockDim.x + threadIdx.x;
  const int total = Bb * Ll * Oo;
  if (tid >= total) return;
  int o = tid & (Oo - 1);
  int l = (tid >> 8) & (Ll - 1);
  int b = tid >> 15;
  float ts = (float)l / (float)(Ll - 1);
  int j = (int)floorf(ts * (float)NSTEPS);
  if (j > NSTEPS - 1) j = NSTEPS - 1;
  if (j < 0) j = 0;
  float gj  = (float)j * 0.05f;
  float gj1 = (j + 1 == NSTEPS) ? 1.0f : (float)(j + 1) * 0.05f;
  float w = (ts - gj) / (gj1 - gj);
  float yj  = y[((size_t)j       * Bb + b) * Oo + o];
  float yj1 = y[((size_t)(j + 1) * Bb + b) * Oo + o];
  out[tid] = yj * (1.f - w) + yj1 * w;
}

// ---------------------------------------------------------------------------
extern "C" void kernel_launch(void* const* d_in, const int* in_sizes, int n_in,
                              void* d_out, int out_size, void* d_ws, size_t ws_size,
                              hipStream_t stream) {
  (void)in_sizes; (void)n_in; (void)out_size; (void)ws_size;
  const float* traj  = (const float*)d_in[0];
  // d_in[1] traj_length: unused by reference
  const float* W_lin = (const float*)d_in[2];
  const float* b_lin = (const float*)d_in[3];
  const float* W_out = (const float*)d_in[4];
  const float* b_out = (const float*)d_in[5];
  const float* W_z0  = (const float*)d_in[6];
  const float* b_z0  = (const float*)d_in[7];
  float* out = (float*)d_out;

  // Workspace carve (~84 MB total)
  char* ws = (char*)d_ws;
  size_t off = 0;
  auto carve = [&](size_t bytes) -> void* {
    void* p = ws + off;
    off = (off + bytes + 255) & ~(size_t)255;
    return p;
  };
  _Float16* Wf16   = (_Float16*)carve((size_t)(Hh * Dd) * Hh * 2); // 75.5 MB, L2-resident
  _Float16* Wout16 = (_Float16*)carve((size_t)Oo * Hh * 2);
  float*    pT     = (float*)carve(Dd * Bb * 4);
  float*    qT     = (float*)carve(Dd * Bb * 4);
  float*    zA32   = (float*)carve(Bb * Hh * 4);
  float*    zB32   = (float*)carve(Bb * Hh * 4);
  _Float16* zA16   = (_Float16*)carve(Bb * Hh * 2);
  _Float16* zB16   = (_Float16*)carve(Bb * Hh * 2);
  float*    zs32   = (float*)carve((size_t)(NSTEPS + 1) * Bb * Hh * 4);
  _Float16* zs16   = (_Float16*)carve((size_t)(NSTEPS + 1) * Bb * Hh * 2);
  float*    ygrid  = (float*)carve((size_t)(NSTEPS + 1) * Bb * Oo * 4);

  // 1) spline p,q
  spline_kernel<<<(Bb * Dd + 255) / 256, 256, 0, stream>>>(traj, pT, qT);

  // 2) weight conversions to f16
  cvt_kernel<<<4096, 256, 0, stream>>>(W_lin, Wf16, (Hh * Dd) * Hh / 4);
  cvt_kernel<<<192, 256, 0, stream>>>(W_out, Wout16, Oo * Hh / 4);

  // 3) z0
  z0_kernel<<<(Bb * Hh + 255) / 256, 256, 0, stream>>>(
      traj, W_z0, b_z0, zA32, zA16, zs32);

  // 4) 20 Euler steps (ping-pong; grid computed exactly as reference, fp32)
  float gridv[NSTEPS + 1];
  for (int k = 0; k <= NSTEPS; ++k) gridv[k] = (float)k * 0.05f;
  gridv[NSTEPS] = 1.0f;
  float *zi32 = zA32, *zo32 = zB32;
  _Float16 *zi16 = zA16, *zo16 = zB16;
  for (int k = 0; k < NSTEPS; ++k) {
    float t = gridv[k], dt = gridv[k + 1] - gridv[k];
    step_kernel<<<(4 * Hh) / 4, 128, 0, stream>>>(   // 3072 waves, 4/block
        zi32, zi16, Wf16, b_lin, pT, qT, t, dt,
        zo32, zo16, zs32 + (size_t)(k + 1) * Bb * Hh);
    float* t32 = zi32; zi32 = zo32; zo32 = t32;
    _Float16* t16 = zi16; zi16 = zo16; zo16 = t16;
  }

  // 5) convert grid states to f16, project through W_out
  cvt_kernel<<<1024, 256, 0, stream>>>(zs32, zs16, (NSTEPS + 1) * Bb * Hh / 4);
  outproj_kernel<<<((NSTEPS + 1) * Bb / 16) * 16 / 4, 128, 0, stream>>>(
      zs16, Wout16, b_out, ygrid);

  // 6) interpolate onto output times
  interp_kernel<<<(Bb * Ll * Oo + 255) / 256, 256, 0, stream>>>(ygrid, out);
}
